// Adapter_63436666962301
// MI455X (gfx1250) — matrix operand order repaired and checked by reading, not compiled, not run on hardware
//
#include <hip/hip_runtime.h>
#include <hip/hip_bf16.h>

// ---------------------------------------------------------------------------
// Shapes (fixed by the reference)
#define B_  32
#define SX  2048
#define SY  512
#define D_  1024
#define T_  2
#define E_  128     // DOWN = 1024/8
#define O_  1024
// M = B*Sx = 65536 rows, tiled in 128-row tiles -> 512 M-tiles (16 per batch)
// ---------------------------------------------------------------------------

typedef __attribute__((ext_vector_type(16))) __bf16       v16bf;
typedef __attribute__((ext_vector_type(8)))  float        v8f;
typedef __attribute__((ext_vector_type(4)))  unsigned int u32x4;

union FragU {
    u32x4 u[2];
    v16bf b;
};

__device__ __forceinline__ unsigned short f2bf(float f) {
    unsigned int u = __float_as_uint(f);
    u += 0x7FFFu + ((u >> 16) & 1u);   // round-to-nearest-even
    return (unsigned short)(u >> 16);
}

// ---------------------------------------------------------------------------
// Kernel 0: convert W_down [E,D] and W_up [O,E] fp32 -> bf16 (stay hot in L2)
// ---------------------------------------------------------------------------
__global__ void k_convert_weights(const float* __restrict__ Wd,
                                  const float* __restrict__ Wu,
                                  unsigned short* __restrict__ Wd16,
                                  unsigned short* __restrict__ Wu16) {
    int idx = blockIdx.x * blockDim.x + threadIdx.x;
    if (idx < E_ * D_) Wd16[idx] = f2bf(Wd[idx]);
    if (idx < O_ * E_) Wu16[idx] = f2bf(Wu[idx]);
}

// ---------------------------------------------------------------------------
// Kernel 1: tokens[b,t,:] = latent[t,:] + softmax_s(latent[t]·y[b,s]) @ y[b]
// One block (256 thr) per batch.
// ---------------------------------------------------------------------------
__global__ void k_tokens(const float* __restrict__ y,
                         const float* __restrict__ latent,
                         float* __restrict__ tokensOut) {
    __shared__ float tokL[T_ * D_];     // 8 KB
    __shared__ float lsm[T_][SY];       // 4 KB (logits, then exp weights)
    __shared__ float red[256];
    __shared__ float invs[T_];
    int tid = threadIdx.x;
    int b   = blockIdx.x;

    for (int i = tid; i < T_ * D_; i += 256) tokL[i] = latent[i];
    __syncthreads();

    // logits l[t][s] = latent[t] . y[b,s]
    for (int s = tid; s < SY; s += 256) {
        const float* yr = y + ((size_t)b * SY + s) * D_;
        float a0 = 0.f, a1 = 0.f;
        for (int d = 0; d < D_; d += 4) {
            float4 v = *(const float4*)(yr + d);
            a0 += v.x * tokL[d]      + v.y * tokL[d + 1]
                + v.z * tokL[d + 2]  + v.w * tokL[d + 3];
            a1 += v.x * tokL[D_ + d]     + v.y * tokL[D_ + d + 1]
                + v.z * tokL[D_ + d + 2] + v.w * tokL[D_ + d + 3];
        }
        lsm[0][s] = a0; lsm[1][s] = a1;
    }
    __syncthreads();

    // softmax over s per t (keep unnormalized exp, fold 1/sum into phase B)
    for (int t = 0; t < T_; ++t) {
        float m = fmaxf(lsm[t][tid], lsm[t][tid + 256]);
        red[tid] = m; __syncthreads();
        for (int o = 128; o > 0; o >>= 1) {
            if (tid < o) red[tid] = fmaxf(red[tid], red[tid + o]);
            __syncthreads();
        }
        float mv = red[0]; __syncthreads();
        float sl = 0.f;
        for (int s = tid; s < SY; s += 256) {
            float e = __expf(lsm[t][s] - mv);
            lsm[t][s] = e; sl += e;
        }
        red[tid] = sl; __syncthreads();
        for (int o = 128; o > 0; o >>= 1) {
            if (tid < o) red[tid] += red[tid + o];
            __syncthreads();
        }
        if (tid == 0) invs[t] = 1.f / red[0];
        __syncthreads();
    }

    // weighted sum over y: each thread owns 4 d-columns, coalesced over s
    float acc0[4] = {0, 0, 0, 0}, acc1[4] = {0, 0, 0, 0};
    for (int s = 0; s < SY; ++s) {
        float a0 = lsm[0][s], a1 = lsm[1][s];
        const float* yr = y + ((size_t)b * SY + s) * D_;
#pragma unroll
        for (int j = 0; j < 4; ++j) {
            float v = yr[tid + 256 * j];
            acc0[j] += a0 * v; acc1[j] += a1 * v;
        }
    }
#pragma unroll
    for (int j = 0; j < 4; ++j) {
        int d = tid + 256 * j;
        tokensOut[((size_t)b * T_ + 0) * D_ + d] = tokL[d]      + invs[0] * acc0[j];
        tokensOut[((size_t)b * T_ + 1) * D_ + d] = tokL[D_ + d] + invs[1] * acc1[j];
    }
}

// ---------------------------------------------------------------------------
// Kernel 2: fused x-attention + gated residual + down-proj + ReLU -> z bf16
// Block = 256 thr (8 waves), 128-row M-tile; z tile is 128x128.
// Register-prefetched, LDS double-buffered: one barrier per k-chunk.
// ---------------------------------------------------------------------------
__global__ void k_attn_down(const float* __restrict__ x,
                            const float* __restrict__ tokens,
                            const float* __restrict__ gate,
                            const unsigned short* __restrict__ Wd16,
                            unsigned short* __restrict__ z16) {
    __shared__ float tokS[T_ * D_];                 // 8 KB
    __shared__ float c0s[128], c1s[128];            // per-row gated coeffs
    __shared__ float p0[256], p1[256];
    __shared__ unsigned short Als[2][128][32];      // 16 KB  A (bf16 x_new)
    __shared__ unsigned short Bls[2][128][32];      // 16 KB  B (bf16 W_down)
    int tid = threadIdx.x;
    size_t mbase = (size_t)blockIdx.x * 128;
    int b = blockIdx.x >> 4;                        // 16 M-tiles per batch

    for (int i = tid; i < T_ * D_; i += 256)
        tokS[i] = tokens[(size_t)b * T_ * D_ + i];
    __syncthreads();

    // ---- phase 1: per-row logits vs tokens, 2 threads per row -------------
    {
        int row = tid >> 1, part = tid & 1;
        const float* xr = x + (mbase + row) * D_ + part * 512;
        const float* t0 = tokS + part * 512;
        const float* t1 = tokS + D_ + part * 512;
        float d0 = 0.f, d1 = 0.f;
        for (int d = 0; d < 512; d += 4) {
            float4 v = *(const float4*)(xr + d);
            d0 += v.x * t0[d] + v.y * t0[d + 1] + v.z * t0[d + 2] + v.w * t0[d + 3];
            d1 += v.x * t1[d] + v.y * t1[d + 1] + v.z * t1[d + 2] + v.w * t1[d + 3];
        }
        p0[tid] = d0; p1[tid] = d1;
        __syncthreads();
        if (part == 0) {
            float l0 = d0 + p0[tid + 1], l1 = d1 + p1[tid + 1];
            float m  = fmaxf(l0, l1);
            float e0 = __expf(l0 - m), e1 = __expf(l1 - m);
            float inv = gate[0] / (e0 + e1);
            c0s[row] = e0 * inv; c1s[row] = e1 * inv;
        }
        __syncthreads();
    }

    // ---- phase 2: z = relu((x + c0*tok0 + c1*tok1) @ W_down^T) ------------
    v8f acc[8] = {};
    int lane = tid & 31, w = tid >> 5;
    int half = lane >> 4, rr = lane & 15;

    float4 xr_[4];      // prefetch regs: 16 floats of x per thread
    u32x4  wr_[2];      // prefetch regs: 16 bf16 of W_down per thread

    auto issue = [&](int kc) {
        int kb = kc * 32;
#pragma unroll
        for (int j = 0; j < 4; ++j) {
            int idx4 = tid + 256 * j;            // 0..1023 float4 slots
            int r    = idx4 >> 3;                // 8 float4 per row
            int k4   = (idx4 & 7) << 2;
            xr_[j] = *(const float4*)(x + (mbase + r) * D_ + kb + k4);
        }
#pragma unroll
        for (int j = 0; j < 2; ++j) {
            int idx8 = tid + 256 * j;            // 0..511 b128 slots
            int e    = idx8 >> 2;                // 4 b128 per row
            int k8   = (idx8 & 3) << 3;
            wr_[j] = *(const u32x4*)(Wd16 + e * D_ + kb + k8);
        }
    };

    issue(0);
    for (int kc = 0; kc < D_ / 32; ++kc) {
        int p  = kc & 1;
        int kb = kc * 32;
        // convert + store current chunk into LDS buffer p
#pragma unroll
        for (int j = 0; j < 4; ++j) {
            int idx4 = tid + 256 * j;
            int r    = idx4 >> 3;
            int k4   = (idx4 & 7) << 2;
            float c0 = c0s[r], c1 = c1s[r];
            float v0 = xr_[j].x + c0 * tokS[kb + k4]     + c1 * tokS[D_ + kb + k4];
            float v1 = xr_[j].y + c0 * tokS[kb + k4 + 1] + c1 * tokS[D_ + kb + k4 + 1];
            float v2 = xr_[j].z + c0 * tokS[kb + k4 + 2] + c1 * tokS[D_ + kb + k4 + 2];
            float v3 = xr_[j].w + c0 * tokS[kb + k4 + 3] + c1 * tokS[D_ + kb + k4 + 3];
            unsigned int lo = (unsigned int)f2bf(v0) | ((unsigned int)f2bf(v1) << 16);
            unsigned int hi = (unsigned int)f2bf(v2) | ((unsigned int)f2bf(v3) << 16);
            unsigned long long q = (unsigned long long)lo | ((unsigned long long)hi << 32);
            *(unsigned long long*)&Als[p][r][k4] = q;
        }
#pragma unroll
        for (int j = 0; j < 2; ++j) {
            int idx8 = tid + 256 * j;
            int e    = idx8 >> 2;
            int k8   = (idx8 & 3) << 3;
            *(u32x4*)&Bls[p][e][k8] = wr_[j];
        }
        // prefetch next chunk (overlaps with WMMA below)
        if (kc + 1 < D_ / 32) issue(kc + 1);
        __syncthreads();

        // A fragment: lane half h holds K = [8h,8h+8) and [8h+16,8h+24)
        FragU aF;
        const unsigned int* ap = (const unsigned int*)(&Als[p][16 * w + rr][0]);
        aF.u[0] = *(const u32x4*)(ap + half * 4);
        aF.u[1] = *(const u32x4*)(ap + half * 4 + 8);
        // preload all 8 B fragments, then issue WMMAs back-to-back
        FragU bF[8];
#pragma unroll
        for (int e = 0; e < 8; ++e) {
            const unsigned int* bp = (const unsigned int*)(&Bls[p][16 * e + rr][0]);
            bF[e].u[0] = *(const u32x4*)(bp + half * 4);
            bF[e].u[1] = *(const u32x4*)(bp + half * 4 + 8);
        }
#pragma unroll
        for (int e = 0; e < 8; ++e)
            acc[e] = __builtin_amdgcn_wmma_f32_16x16x32_bf16(
                false, aF.b, false, bF[e].b, (short)0, acc[e], false, false);
        // no trailing barrier: next iter writes the other LDS buffer
    }

    // ---- ReLU + store z as bf16 ------------------------------------------
#pragma unroll
    for (int e = 0; e < 8; ++e) {
#pragma unroll
        for (int r = 0; r < 8; ++r) {
            float v = fmaxf(acc[e][r], 0.f);
            size_t grow = mbase + 16 * w + r + half * 8;
            int gcol = e * 16 + rr;
            z16[grow * E_ + gcol] = f2bf(v);
        }
    }
}

// ---------------------------------------------------------------------------
// Kernel 3: out = z @ W_up^T.  Grid (512 Mtiles, 8 Ntiles), 128x128 out tile.
// Full K=128 staged in LDS (64 KB).
// ---------------------------------------------------------------------------
__global__ void k_up(const unsigned short* __restrict__ z16,
                     const unsigned short* __restrict__ Wu16,
                     float* __restrict__ out) {
    __shared__ unsigned short Zls[128][128];   // 32 KB
    __shared__ unsigned short Uls[128][128];   // 32 KB
    int tid = threadIdx.x;
    size_t mbase = (size_t)blockIdx.x * 128;
    int nb = blockIdx.y * 128;

    const u32x4* zsrc = (const u32x4*)(z16 + mbase * E_);
    const u32x4* usrc = (const u32x4*)(Wu16 + (size_t)nb * E_);
    u32x4* zdst = (u32x4*)&Zls[0][0];
    u32x4* udst = (u32x4*)&Uls[0][0];
    for (int i = tid; i < 128 * 128 / 8; i += 256) {   // 2048 b128 copies
        zdst[i] = zsrc[i];
        udst[i] = usrc[i];
    }
    __syncthreads();

    int lane = tid & 31, w = tid >> 5;
    int half = lane >> 4, rr = lane & 15;
    v8f acc[8] = {};
#pragma unroll
    for (int kc = 0; kc < 4; ++kc) {
        int kofs = kc * 16;  // dwords: kc*32 ushorts
        FragU aF;
        const unsigned int* ap = (const unsigned int*)(&Zls[16 * w + rr][0]);
        aF.u[0] = *(const u32x4*)(ap + kofs + half * 4);
        aF.u[1] = *(const u32x4*)(ap + kofs + half * 4 + 8);
        FragU bF[8];
#pragma unroll
        for (int n = 0; n < 8; ++n) {
            const unsigned int* bp = (const unsigned int*)(&Uls[16 * n + rr][0]);
            bF[n].u[0] = *(const u32x4*)(bp + kofs + half * 4);
            bF[n].u[1] = *(const u32x4*)(bp + kofs + half * 4 + 8);
        }
#pragma unroll
        for (int n = 0; n < 8; ++n)
            acc[n] = __builtin_amdgcn_wmma_f32_16x16x32_bf16(
                false, aF.b, false, bF[n].b, (short)0, acc[n], false, false);
    }
#pragma unroll
    for (int n = 0; n < 8; ++n)
#pragma unroll
        for (int r = 0; r < 8; ++r)
            out[(mbase + 16 * w + r + half * 8) * O_ + nb + n * 16 + rr] = acc[n][r];
}

// ---------------------------------------------------------------------------
extern "C" void kernel_launch(void* const* d_in, const int* in_sizes, int n_in,
                              void* d_out, int out_size, void* d_ws, size_t ws_size,
                              hipStream_t stream) {
    const float* x      = (const float*)d_in[0];
    const float* y      = (const float*)d_in[1];
    const float* latent = (const float*)d_in[2];
    const float* gate   = (const float*)d_in[3];
    const float* Wd     = (const float*)d_in[4];
    const float* Wu     = (const float*)d_in[5];
    float* out = (float*)d_out;

    char* ws = (char*)d_ws;
    float*          tokensWS = (float*)ws;                           // 256 KB
    unsigned short* Wd16     = (unsigned short*)(ws + (256 << 10));  // 256 KB
    unsigned short* Wu16     = (unsigned short*)(ws + (512 << 10));  // 256 KB
    unsigned short* z16      = (unsigned short*)(ws + (768 << 10));  // 16 MB

    k_convert_weights<<<512, 256, 0, stream>>>(Wd, Wu, Wd16, Wu16);
    k_tokens<<<B_, 256, 0, stream>>>(y, latent, tokensWS);
    k_attn_down<<<(B_ * SX) / 128, 256, 0, stream>>>(x, tokensWS, gate, Wd16, z16);
    k_up<<<dim3((B_ * SX) / 128, O_ / 128), 256, 0, stream>>>(z16, Wu16, out);
}
